// PointTransformerDecoderLDIF_39565238730826
// MI455X (gfx1250) — matrix operand order repaired and checked by reading, not compiled, not run on hardware
//
#include <hip/hip_runtime.h>
#include <hip/hip_bf16.h>

typedef __attribute__((ext_vector_type(16))) _Float16     v16h;
typedef __attribute__((ext_vector_type(8)))  float        v8f;
typedef __attribute__((ext_vector_type(4)))  unsigned int v4u;

constexpr int Bn  = 2;
constexpr int Qn  = 4096;
constexpr int An  = 32;
constexpr int DIN = 128;
constexpr int DIM = 128;
constexpr int HID = 64;
constexpr int NBk = 5;

// ---------------- workspace (d_ws) layout, bytes ----------------
constexpr int WS_GEOM = 0;       // 64 anchors * 16 f32   = 4096
constexpr int WS_W0F  = 4096;    // 64 * 128 f32          = 32768
constexpr int WS_WCA  = 36864;   // 64 * 5 * 64 f32       = 81920
constexpr int WS_WCL  = 118784;  // 5 * 64 * 3 f32        = 3840 (pad to 4096)
constexpr int WS_W1H  = 122880;  // 64*128 f16            = 16384
constexpr int WS_WB0H = 139264;  // 5*64*64 f16           = 40960
constexpr int WS_WB1H = 180224;  // 5*64*64 f16           = 40960  (end 221184)

// ---------------- dynamic LDS layout, bytes ----------------
constexpr int LDS_W0F   = 0;       // 32*128 f32 = 16384
constexpr int LDS_WCA   = 16384;   // 32*5*64 f32 = 40960
constexpr int LDS_GEOM  = 57344;   // 32*16 f32 = 2048
constexpr int LDS_W0LOC = 59392;   // 128 float4 = 2048
constexpr int LDS_WCL   = 61440;   // 5*64 float4 = 5120
constexpr int LDS_B1    = 66560;   // 64 f32 = 256
constexpr int LDS_BB0   = 66816;   // 320 f32 = 1280
constexpr int LDS_BB1   = 68096;   // 320 f32 = 1280
constexpr int LDS_WOUT  = 69376;   // 64 f32 = 256
constexpr int LDS_QBUF  = 69632;   // 16 float4 = 256
constexpr int LDS_WPART = 69888;   // 8 waves * 16 f32 = 512
constexpr int LDS_LOC   = 70400;   // 8 waves * 16 float4 = 2048
constexpr int LDS_W1H   = 72448;   // f16 blob W1|Wb0|Wb1 = 98304 (contiguous)
constexpr int LDS_HBUF  = 170752;  // 8 waves * 16*64 f16 = 16384
constexpr int LDS_TOTAL = 187136;

// =====================================================================
// Prep 1: per-anchor tables (rot, inv-diag, weight scale, W0f, WcA)
// =====================================================================
__global__ __launch_bounds__(128) void ldif_prep_anchor(
    const float* __restrict__ anchors, const float* __restrict__ feats,
    const float* __restrict__ Ws, const float* __restrict__ bs,
    const float* __restrict__ Wr, const float* __restrict__ br,
    const float* __restrict__ W0, const float* __restrict__ b0,
    const float* __restrict__ Wc, const float* __restrict__ bc,
    char* __restrict__ ws)
{
    __shared__ float fsh[DIN];
    __shared__ float w0f[DIM];
    __shared__ float sr[6];
    const int tid = threadIdx.x;
    const int ba  = blockIdx.x;               // b*An + a

    fsh[tid] = feats[(size_t)ba * DIN + tid];
    __syncthreads();

    if (tid < 6) {
        const int j = tid % 3;
        const float* row = (tid < 3 ? Ws : Wr) + j * DIN;
        float s = (tid < 3 ? bs[j] : br[j]);
        for (int c = 0; c < DIN; ++c) s += row[c] * fsh[c];
        sr[tid] = s;
    }
    {   // W0f = W0[:, :128] @ f + b0
        float s = b0[tid];
        const float* row = W0 + (size_t)tid * (DIN + 3);
        for (int c = 0; c < DIN; ++c) s += row[c] * fsh[c];
        w0f[tid] = s;
        ((float*)(ws + WS_W0F))[(size_t)ba * DIM + tid] = s;
    }
    __syncthreads();

    if (tid < HID) {  // WcA[i] = Wc[i] @ W0f + bc[i]
        for (int i = 0; i < NBk; ++i) {
            float s = bc[i * HID + tid];
            const float* row = Wc + (size_t)(i * HID + tid) * DIM;
            for (int c = 0; c < DIM; ++c) s += row[c] * w0f[c];
            ((float*)(ws + WS_WCA))[(size_t)ba * NBk * HID + i * HID + tid] = s;
        }
    }
    if (tid == 0) {
        float* g = (float*)(ws + WS_GEOM) + ba * 16;
        const float d0 = 0.005f + 1.f / (1.f + __expf(-sr[0]));
        const float d1 = 0.005f + 1.f / (1.f + __expf(-sr[1]));
        const float d2 = 0.005f + 1.f / (1.f + __expf(-sr[2]));
        const float TP = 6.2831853071795864f;
        const float ax = TP / (1.f + __expf(-sr[3]));
        const float ay = TP / (1.f + __expf(-sr[4]));
        const float az = TP / (1.f + __expf(-sr[5]));
        const float cx = __cosf(ax), sx = __sinf(ax);
        const float cy = __cosf(ay), sy = __sinf(ay);
        const float cz = __cosf(az), sz = __sinf(az);
        // R = Rz @ Ry @ Rx (row-major)
        g[0] = cz * cy; g[1] = -sz * cx + cz * sy * sx; g[2] = sz * sx + cz * sy * cx;
        g[3] = sz * cy; g[4] =  cz * cx + sz * sy * sx; g[5] = -cz * sx + sz * sy * cx;
        g[6] = -sy;     g[7] =  cy * sx;                g[8] = cy * cx;
        g[9]  = 1.f / d0; g[10] = 1.f / d1; g[11] = 1.f / d2;
        g[12] = (d0 * d1 * d2) * (1.f / 248.05021344239853f);  // cov_det/(2pi)^3
        g[13] = anchors[ba * 3 + 0];
        g[14] = anchors[ba * 3 + 1];
        g[15] = anchors[ba * 3 + 2];
    }
}

// =====================================================================
// Prep 2: f16 weight blob (row-major [N][K] => contiguous B-fragments)
//         and WcL[i] = Wc[i] @ W0loc  ([5][64][3])
// =====================================================================
__global__ __launch_bounds__(256) void ldif_prep_weights(
    const float* __restrict__ W0, const float* __restrict__ W1,
    const float* __restrict__ Wc, const float* __restrict__ Wb0,
    const float* __restrict__ Wb1, char* __restrict__ ws)
{
    const int gid    = blockIdx.x * 256 + threadIdx.x;
    const int stride = gridDim.x * 256;
    _Float16* w1h  = (_Float16*)(ws + WS_W1H);
    _Float16* wb0h = (_Float16*)(ws + WS_WB0H);
    _Float16* wb1h = (_Float16*)(ws + WS_WB1H);
    for (int i = gid; i < HID * DIM;       i += stride) w1h[i]  = (_Float16)W1[i];
    for (int i = gid; i < NBk * HID * HID; i += stride) wb0h[i] = (_Float16)Wb0[i];
    for (int i = gid; i < NBk * HID * HID; i += stride) wb1h[i] = (_Float16)Wb1[i];

    float* wcl = (float*)(ws + WS_WCL);
    for (int j = gid; j < NBk * HID * 3; j += stride) {
        const int i   = j / (HID * 3);
        const int rem = j % (HID * 3);
        const int nn  = rem / 3, jj = rem % 3;
        const float* row = Wc + (size_t)(i * HID + nn) * DIM;
        float s = 0.f;
        for (int c = 0; c < DIM; ++c) s += row[c] * W0[(size_t)c * (DIN + 3) + DIN + jj];
        wcl[j] = s;
    }
}

// =====================================================================
// Main: 1 WG = 16 queries; 8 waves x 4 anchors; WMMA f16 MLP chain.
// Weight blob staged global->LDS with GLOBAL_LOAD_ASYNC_TO_LDS_B128.
// =====================================================================
__global__ __launch_bounds__(256) void ldif_main(
    const float* __restrict__ xyz_q, const float* __restrict__ W0,
    const float* __restrict__ b1,   const float* __restrict__ bb0,
    const float* __restrict__ bb1,  const float* __restrict__ Wout,
    const float* __restrict__ bout, const char* __restrict__ ws,
    float* __restrict__ out)
{
    extern __shared__ char smem[];
    const int tid = threadIdx.x;
    const int b   = blockIdx.x >> 8;          // Qn/16 = 256 tiles per batch
    const int q0  = (blockIdx.x & 255) << 4;

    float*    lW0f  = (float*)(smem + LDS_W0F);
    float*    lWcA  = (float*)(smem + LDS_WCA);
    float*    lGeom = (float*)(smem + LDS_GEOM);
    float4*   lW0l  = (float4*)(smem + LDS_W0LOC);
    float4*   lWcL  = (float4*)(smem + LDS_WCL);
    float*    lB1   = (float*)(smem + LDS_B1);
    float*    lBB0  = (float*)(smem + LDS_BB0);
    float*    lBB1  = (float*)(smem + LDS_BB1);
    float*    lWout = (float*)(smem + LDS_WOUT);
    float4*   lQ    = (float4*)(smem + LDS_QBUF);
    float*    lWp   = (float*)(smem + LDS_WPART);
    float4*   lLoc  = (float4*)(smem + LDS_LOC);
    _Float16* lW1h  = (_Float16*)(smem + LDS_W1H);
    _Float16* lWb0h = (_Float16*)(smem + LDS_W1H + 16384);
    _Float16* lWb1h = (_Float16*)(smem + LDS_W1H + 16384 + 40960);
    _Float16* lHb   = (_Float16*)(smem + LDS_HBUF);

    // L2-resident tables: prefetch the hot per-anchor region early.
    __builtin_prefetch(ws + WS_GEOM + (size_t)b * An * 16 * 4, 0, 1);

    // ---- async DMA: f16 weight blob (W1|Wb0|Wb1, 98304 B) global -> LDS ----
    // GVS mode: %0 = LDS byte offset VGPR (dynamic LDS starts at offset 0),
    //           %1 = 32-bit global offset VGPR, %2 = SGPR-pair base address.
    // Each lane moves 16 B per issue; tracked on ASYNCcnt.
    {
        const char* src = ws + WS_W1H;
#pragma unroll
        for (int it = 0; it < (16384 + 40960 + 40960) / (256 * 16); ++it) {
            const unsigned off = (unsigned)((it * 256 + tid) * 16);
            const unsigned lds = (unsigned)LDS_W1H + off;
            asm volatile("global_load_async_to_lds_b128 %0, %1, %2"
                         :: "v"(lds), "v"(off), "s"(src)
                         : "memory");
        }
    }

    // ---- stage f32 tables for this batch into LDS (regular path) ----
    const float* wsf = (const float*)ws;
    for (int i = tid; i < An * DIM;        i += 256) lW0f[i]  = wsf[WS_W0F  / 4 + b * An * DIM + i];
    for (int i = tid; i < An * NBk * HID;  i += 256) lWcA[i]  = wsf[WS_WCA  / 4 + b * An * NBk * HID + i];
    for (int i = tid; i < An * 16;         i += 256) lGeom[i] = wsf[WS_GEOM / 4 + b * An * 16 + i];
    if (tid < DIM) {
        const float* r = W0 + (size_t)tid * (DIN + 3) + DIN;
        lW0l[tid] = make_float4(r[0], r[1], r[2], 0.f);
    }
    for (int i = tid; i < NBk * HID; i += 256) {
        const float* r = wsf + WS_WCL / 4 + i * 3;
        lWcL[i] = make_float4(r[0], r[1], r[2], 0.f);
        lBB0[i] = bb0[i];
        lBB1[i] = bb1[i];
    }
    if (tid < HID) { lB1[tid] = b1[tid]; lWout[tid] = Wout[tid]; }
    if (tid < 16) {
        const float* q = xyz_q + ((size_t)b * Qn + q0 + tid) * 3;
        lQ[tid] = make_float4(q[0], q[1], q[2], 0.f);
    }
    // Drain this wave's async LDS writes, then barrier so every wave sees
    // the full weight blob.
    asm volatile("s_wait_asynccnt 0x0" ::: "memory");
    __syncthreads();

    const int lane = tid & 31;
    const int wave = tid >> 5;
    const int n    = lane & 15;      // N / A-row index
    const int hi   = lane >> 4;
    const float boutv = bout[0];
    _Float16* hb = lHb + wave * 16 * 64;

    float qpart[8];
#pragma unroll
    for (int v = 0; v < 8; ++v) qpart[v] = 0.f;

    for (int ai = 0; ai < 4; ++ai) {
        const int a = wave * 4 + ai;
        const float* g = lGeom + a * 16;

        // per-row Gaussian geometry (this lane owns A-row M = n)
        const float4 qv = lQ[n];
        const float dx = g[13] - qv.x, dy = g[14] - qv.y, dz = g[15] - qv.z;
        const float m  = dx * dx * g[9] + dy * dy * g[10] + dz * dz * g[11];
        const float wgt = __expf(-0.5f * m) * g[12];
        const float lx = g[0] * dx + g[1] * dy + g[2] * dz;
        const float ly = g[3] * dx + g[4] * dy + g[5] * dz;
        const float lz = g[6] * dx + g[7] * dy + g[8] * dz;
        if (hi == 0) lLoc[wave * 16 + n] = make_float4(lx, ly, lz, wgt);
        float4 locM[8];
#pragma unroll
        for (int v = 0; v < 8; ++v) locM[v] = lLoc[wave * 16 + hi * 8 + v];  // C-rows M=v+8*hi

        // ---- layer 1: net = W1 @ relu(lat) + b1, lat built in registers ----
        const float* w0fa = lW0f + a * DIM;
        v8f netA[4];
#pragma unroll
        for (int t = 0; t < 4; ++t) {
            const float bv = lB1[t * 16 + n];
            v8f x;
#pragma unroll
            for (int v = 0; v < 8; ++v) x[v] = bv;
            netA[t] = x;
        }
#pragma unroll
        for (int kk = 0; kk < 4; ++kk) {
            v16h at;
            const int base = kk * 32 + hi * 16;
#pragma unroll
            for (int h = 0; h < 8; ++h) {
                const int c0 = base + h, c1 = base + 8 + h;
                const float4 w0v = lW0l[c0], w1v = lW0l[c1];
                const float l0 = w0fa[c0] + w0v.x * lx + w0v.y * ly + w0v.z * lz;
                const float l1 = w0fa[c1] + w1v.x * lx + w1v.y * ly + w1v.z * lz;
                at[h]     = (_Float16)fmaxf(l0, 0.f);
                at[h + 8] = (_Float16)fmaxf(l1, 0.f);
            }
#pragma unroll
            for (int t = 0; t < 4; ++t) {
                const v16h bt = *(const v16h*)(lW1h + (t * 16 + n) * DIM + kk * 32 + hi * 16);
                netA[t] = __builtin_amdgcn_wmma_f32_16x16x32_f16(
                    false, at, false, bt, (short)0, netA[t], false, false);
            }
        }

        // ---- 5 residual blocks ----
        for (int i = 0; i < NBk; ++i) {
            // linear injection: net += WcA[i][a] + WcL[i] @ loc
#pragma unroll
            for (int t = 0; t < 4; ++t) {
                const int N = t * 16 + n;
                const float  wca = lWcA[a * NBk * HID + i * HID + N];
                const float4 wl  = lWcL[i * HID + N];
#pragma unroll
                for (int v = 0; v < 8; ++v)
                    netA[t][v] += wca + wl.x * locM[v].x + wl.y * locM[v].y + wl.z * locM[v].z;
            }
            // relu(net) -> LDS (C layout -> row-major [M][N] halfs)
#pragma unroll
            for (int t = 0; t < 4; ++t)
#pragma unroll
                for (int v = 0; v < 8; ++v)
                    hb[(v + 8 * hi) * 64 + t * 16 + n] = (_Float16)fmaxf(netA[t][v], 0.f);

            // h = Wb0[i] @ relu(net) + bb0[i]
            v8f hA[4];
#pragma unroll
            for (int t = 0; t < 4; ++t) {
                const float bv = lBB0[i * HID + t * 16 + n];
                v8f x;
#pragma unroll
                for (int v = 0; v < 8; ++v) x[v] = bv;
                hA[t] = x;
            }
#pragma unroll
            for (int kk = 0; kk < 2; ++kk) {
                union { v16h h; v4u u[2]; } av;
                av.u[0] = *(const v4u*)(hb + n * 64 + kk * 32 + hi * 16);
                av.u[1] = *(const v4u*)(hb + n * 64 + kk * 32 + hi * 16 + 8);
#pragma unroll
                for (int t = 0; t < 4; ++t) {
                    const v16h bt = *(const v16h*)(lWb0h + i * HID * HID + (t * 16 + n) * HID + kk * 32 + hi * 16);
                    hA[t] = __builtin_amdgcn_wmma_f32_16x16x32_f16(
                        false, av.h, false, bt, (short)0, hA[t], false, false);
                }
            }
            // relu(h) -> LDS
#pragma unroll
            for (int t = 0; t < 4; ++t)
#pragma unroll
                for (int v = 0; v < 8; ++v)
                    hb[(v + 8 * hi) * 64 + t * 16 + n] = (_Float16)fmaxf(hA[t][v], 0.f);

            // net += Wb1[i] @ relu(h) + bb1[i]
#pragma unroll
            for (int kk = 0; kk < 2; ++kk) {
                union { v16h h; v4u u[2]; } av;
                av.u[0] = *(const v4u*)(hb + n * 64 + kk * 32 + hi * 16);
                av.u[1] = *(const v4u*)(hb + n * 64 + kk * 32 + hi * 16 + 8);
#pragma unroll
                for (int t = 0; t < 4; ++t) {
                    const v16h bt = *(const v16h*)(lWb1h + i * HID * HID + (t * 16 + n) * HID + kk * 32 + hi * 16);
                    netA[t] = __builtin_amdgcn_wmma_f32_16x16x32_f16(
                        false, av.h, false, bt, (short)0, netA[t], false, false);
                }
            }
#pragma unroll
            for (int t = 0; t < 4; ++t) {
                const float bv = lBB1[i * HID + t * 16 + n];
#pragma unroll
                for (int v = 0; v < 8; ++v) netA[t][v] += bv;
            }
        }

        // ---- head: occ = Wout @ relu(net) + bout; accumulate occ*weight ----
        float part[8];
#pragma unroll
        for (int v = 0; v < 8; ++v) part[v] = 0.f;
#pragma unroll
        for (int t = 0; t < 4; ++t) {
            const float wo = lWout[t * 16 + n];
#pragma unroll
            for (int v = 0; v < 8; ++v) part[v] += fmaxf(netA[t][v], 0.f) * wo;
        }
#pragma unroll
        for (int v = 0; v < 8; ++v) {
            float p = part[v];
            p += __shfl_xor(p, 1, 32);
            p += __shfl_xor(p, 2, 32);
            p += __shfl_xor(p, 4, 32);
            p += __shfl_xor(p, 8, 32);
            qpart[v] += (p + boutv) * locM[v].w;
        }
    }

    // deterministic cross-wave reduction
    if (n == 0) {
#pragma unroll
        for (int v = 0; v < 8; ++v) lWp[wave * 16 + hi * 8 + v] = qpart[v];
    }
    __syncthreads();
    if (tid < 16) {
        float s = 0.f;
#pragma unroll
        for (int w = 0; w < 8; ++w) s += lWp[w * 16 + tid];
        out[(size_t)b * Qn + q0 + tid] = s;
    }
}

// =====================================================================
extern "C" void kernel_launch(void* const* d_in, const int* in_sizes, int n_in,
                              void* d_out, int out_size, void* d_ws, size_t ws_size,
                              hipStream_t stream) {
    (void)in_sizes; (void)n_in; (void)out_size; (void)ws_size;
    const float* xyz_q   = (const float*)d_in[0];
    const float* anchors = (const float*)d_in[1];
    const float* feats   = (const float*)d_in[2];
    const float* Ws   = (const float*)d_in[3];
    const float* bs   = (const float*)d_in[4];
    const float* Wr   = (const float*)d_in[5];
    const float* br   = (const float*)d_in[6];
    const float* W0   = (const float*)d_in[7];
    const float* b0   = (const float*)d_in[8];
    const float* W1   = (const float*)d_in[9];
    const float* b1   = (const float*)d_in[10];
    const float* Wc   = (const float*)d_in[11];
    const float* bc   = (const float*)d_in[12];
    const float* Wb0  = (const float*)d_in[13];
    const float* bb0  = (const float*)d_in[14];
    const float* Wb1  = (const float*)d_in[15];
    const float* bb1  = (const float*)d_in[16];
    const float* Wout = (const float*)d_in[17];
    const float* bout = (const float*)d_in[18];
    char* ws   = (char*)d_ws;
    float* out = (float*)d_out;

    ldif_prep_anchor<<<Bn * An, 128, 0, stream>>>(anchors, feats, Ws, bs, Wr, br,
                                                  W0, b0, Wc, bc, ws);
    ldif_prep_weights<<<8, 256, 0, stream>>>(W0, W1, Wc, Wb0, Wb1, ws);
    ldif_main<<<Bn * (Qn / 16), 256, LDS_TOTAL, stream>>>(xyz_q, W0, b1, bb0, bb1,
                                                          Wout, bout, ws, out);
}
